// LoRADynamicsModel_35656818491677
// MI455X (gfx1250) — matrix operand order repaired and checked by reading, not compiled, not run on hardware
//
#include <hip/hip_runtime.h>

// ---------------- types ----------------
typedef __attribute__((ext_vector_type(16))) __bf16 v16bf;
typedef __attribute__((ext_vector_type(8)))  __bf16 v8bf;
typedef __attribute__((ext_vector_type(8)))  float  v8f;
typedef __attribute__((ext_vector_type(4)))  float  v4f;

#define LORA_SCALE 2.0f

// ---------------- geometry ----------------
static constexpr int IN0  = 896;   // S + A
static constexpr int HDIM = 256;
static constexpr int ODIM = 768;
static constexpr int SDIM = 768;
static constexpr int ADIM = 128;

static constexpr int BM   = 64;    // batch rows per workgroup
static constexpr int NTH  = 256;   // 8 waves (wave32)
static constexpr int KP   = 40;    // padded halves per 32-K panel row (80B, bank-skewed)
static constexpr int HP   = 264;   // padded pitch (halves) for h0/h1 activation tiles

// LDS layout (bytes)
static constexpr int AS_SZ  = BM * KP * 2;            // 5120 per buffer
static constexpr int BS_SZ  = 256 * KP * 2;           // 20480 per buffer
static constexpr int AS_OFF = 0;
static constexpr int BS_OFF = AS_OFF + 2 * AS_SZ;     // 10240
static constexpr int H0_OFF = BS_OFF + 2 * BS_SZ;     // 51200
static constexpr int H1_OFF = H0_OFF + BM * HP * 2;   // 84992
static constexpr int LDS_TOTAL = H1_OFF + BM * HP * 2;// 118784 (< 320KB WGP LDS)

// ---------------- LoRA fold: W_eff = W + scale * U @ D  (bf16 out) ----------------
__global__ void fuse_lora_weights(const float* __restrict__ W,
                                  const float* __restrict__ U,
                                  const float* __restrict__ D,
                                  __bf16* __restrict__ out, int N, int K) {
    int i = blockIdx.x * blockDim.x + threadIdx.x;
    if (i >= N * K) return;
    int n = i / K;
    int k = i - n * K;
    float s = 0.0f;
#pragma unroll
    for (int r = 0; r < 8; ++r) s += U[n * 8 + r] * D[r * K + k];
    out[i] = (__bf16)(W[i] + LORA_SCALE * s);
}

// ---------------- fragment helpers (ISA §7.12.2 layouts, wave32) ----------------
__device__ __forceinline__ v8f wmma_bf16(v16bf a, v16bf b, v8f c) {
    return __builtin_amdgcn_wmma_f32_16x16x32_bf16(false, a, false, b, (short)0, c,
                                                   false, false);
}

// A 16x32 bf16: lane<16 -> row=lane, K={0..7,16..23}; lane>=16 -> row=lane-16, K={8..15,24..31}
__device__ __forceinline__ v16bf load_a(const __bf16* base, int pitch, int row0,
                                        int k0, int lane) {
    int r  = row0 + (lane & 15);
    int kb = (lane < 16) ? 0 : 8;
    const __bf16* p = base + r * pitch + k0 + kb;
    v8bf lo = *(const v8bf*)p;          // K = kb .. kb+7
    v8bf hi = *(const v8bf*)(p + 16);   // K = kb+16 .. kb+23
    v16bf v;
#pragma unroll
    for (int i = 0; i < 8; ++i) { v[i] = lo[i]; v[i + 8] = hi[i]; }
    return v;
}

// B 32x16 bf16 staged in LDS as [n][KP]: lane n=lane&15, K={0..7,16..23} (lane<16) or {8..15,24..31}
__device__ __forceinline__ v16bf load_b(const __bf16* bs, int ntile, int lane) {
    int n  = ntile * 16 + (lane & 15);
    int kb = (lane < 16) ? 0 : 8;
    const __bf16* p = bs + n * KP + kb;
    v8bf lo = *(const v8bf*)p;
    v8bf hi = *(const v8bf*)(p + 16);
    v16bf v;
#pragma unroll
    for (int i = 0; i < 8; ++i) { v[i] = lo[i]; v[i + 8] = hi[i]; }
    return v;
}

// ---------------- staging ----------------
// Stage 64 rows x 32 K of x = concat(state, action) as bf16 (8 floats / thread)
__device__ __forceinline__ void stage_a(__bf16* as, const float* __restrict__ state,
                                        const float* __restrict__ action,
                                        int rowbase, int k0, int tid) {
    int row  = tid >> 2;          // 0..63
    int kk   = (tid & 3) * 8;     // 0,8,16,24
    int kcol = k0 + kk;
    const float* src = (kcol < SDIM)
        ? (state  + (size_t)(rowbase + row) * SDIM + kcol)
        : (action + (size_t)(rowbase + row) * ADIM + (kcol - SDIM));
    v4f f0 = *(const v4f*)src;
    v4f f1 = *(const v4f*)(src + 4);
    v8bf o;
#pragma unroll
    for (int i = 0; i < 4; ++i) { o[i] = (__bf16)f0[i]; o[i + 4] = (__bf16)f1[i]; }
    *(v8bf*)(as + row * KP + kk) = o;
}

// Stage a 256(n) x 32(k) weight panel from W_eff (row-major [N][K]) into LDS [n][KP]
__device__ __forceinline__ void stage_b(__bf16* bs, const __bf16* __restrict__ w,
                                        int K, int nbase, int k0, int tid) {
    const __bf16* src = w + (size_t)(nbase + tid) * K + k0;
    __bf16* dst = bs + tid * KP;
    v8bf v0 = *(const v8bf*)(src);
    v8bf v1 = *(const v8bf*)(src + 8);
    v8bf v2 = *(const v8bf*)(src + 16);
    v8bf v3 = *(const v8bf*)(src + 24);
    *(v8bf*)(dst)      = v0;
    *(v8bf*)(dst + 8)  = v1;
    *(v8bf*)(dst + 16) = v2;
    *(v8bf*)(dst + 24) = v3;
}

// ---------------- fused 3-layer LoRA MLP ----------------
__global__ __launch_bounds__(NTH)
void lora_mlp_fused(const float* __restrict__ state, const float* __restrict__ action,
                    const __bf16* __restrict__ W0e, const float* __restrict__ bias0,
                    const __bf16* __restrict__ W1e, const float* __restrict__ bias1,
                    const __bf16* __restrict__ W2e, const float* __restrict__ bias2,
                    float* __restrict__ out) {
    extern __shared__ char smem[];
    __bf16* As = (__bf16*)(smem + AS_OFF);
    __bf16* Bs = (__bf16*)(smem + BS_OFF);
    __bf16* h0 = (__bf16*)(smem + H0_OFF);
    __bf16* h1 = (__bf16*)(smem + H1_OFF);

    const int tid  = threadIdx.x;
    const int lane = tid & 31;
    const int wave = tid >> 5;
    const int mg   = wave & 3;   // 16-row group within the 64-row tile
    const int nh   = wave >> 1 >> 1; // n-half: 0 or 1 (128 cols each)
    const int rowbase = blockIdx.x * BM;

    const int ASH = AS_SZ / 2;   // halves per A buffer
    const int BSH = BS_SZ / 2;   // halves per B buffer
    const v8f vzero = {0, 0, 0, 0, 0, 0, 0, 0};

    v8f acc[8];
    const int rloc = mg * 16 + ((lane < 16) ? 0 : 8);
    const int nlo  = lane & 15;

    // ================= layer 0: [64,896] x [896,256] =================
#pragma unroll
    for (int t = 0; t < 8; ++t) acc[t] = vzero;
    int cur = 0;
    stage_a(As, state, action, rowbase, 0, tid);
    stage_b(Bs, W0e, IN0, 0, 0, tid);
    __syncthreads();
    for (int ks = 0; ks < IN0 / 32; ++ks) {
        if (ks + 1 < IN0 / 32) {
            stage_a(As + (1 - cur) * ASH, state, action, rowbase, (ks + 1) * 32, tid);
            stage_b(Bs + (1 - cur) * BSH, W0e, IN0, 0, (ks + 1) * 32, tid);
        }
        v16bf a = load_a(As + cur * ASH, KP, mg * 16, 0, lane);
#pragma unroll
        for (int t = 0; t < 8; ++t) {
            v16bf b = load_b(Bs + cur * BSH, nh * 8 + t, lane);
            acc[t] = wmma_bf16(a, b, acc[t]);
        }
        __syncthreads();
        cur ^= 1;
    }
#pragma unroll
    for (int t = 0; t < 8; ++t) {
        int n = nh * 128 + t * 16 + nlo;
        float bv = bias0[n];
#pragma unroll
        for (int j = 0; j < 8; ++j) {
            float v = acc[t][j] + bv;
            h0[(rloc + j) * HP + n] = (__bf16)fmaxf(v, 0.0f);
        }
    }
    __syncthreads();

    // ================= layer 1: [64,256] x [256,256] =================
#pragma unroll
    for (int t = 0; t < 8; ++t) acc[t] = vzero;
    cur = 0;
    stage_b(Bs, W1e, HDIM, 0, 0, tid);
    __syncthreads();
    for (int ks = 0; ks < HDIM / 32; ++ks) {
        if (ks + 1 < HDIM / 32)
            stage_b(Bs + (1 - cur) * BSH, W1e, HDIM, 0, (ks + 1) * 32, tid);
        v16bf a = load_a(h0, HP, mg * 16, ks * 32, lane);
#pragma unroll
        for (int t = 0; t < 8; ++t) {
            v16bf b = load_b(Bs + cur * BSH, nh * 8 + t, lane);
            acc[t] = wmma_bf16(a, b, acc[t]);
        }
        __syncthreads();
        cur ^= 1;
    }
#pragma unroll
    for (int t = 0; t < 8; ++t) {
        int n = nh * 128 + t * 16 + nlo;
        float bv = bias1[n];
#pragma unroll
        for (int j = 0; j < 8; ++j) {
            float v = acc[t][j] + bv;
            h1[(rloc + j) * HP + n] = (__bf16)fmaxf(v, 0.0f);
        }
    }
    __syncthreads();

    // ================= layer 2: [64,256] x [256,768], 3 n-chunks =================
    for (int c = 0; c < 3; ++c) {
#pragma unroll
        for (int t = 0; t < 8; ++t) acc[t] = vzero;
        cur = 0;
        stage_b(Bs, W2e, HDIM, c * 256, 0, tid);
        __syncthreads();
        for (int ks = 0; ks < HDIM / 32; ++ks) {
            if (ks + 1 < HDIM / 32)
                stage_b(Bs + (1 - cur) * BSH, W2e, HDIM, c * 256, (ks + 1) * 32, tid);
            v16bf a = load_a(h1, HP, mg * 16, ks * 32, lane);
#pragma unroll
            for (int t = 0; t < 8; ++t) {
                v16bf b = load_b(Bs + cur * BSH, nh * 8 + t, lane);
                acc[t] = wmma_bf16(a, b, acc[t]);
            }
            __syncthreads();
            cur ^= 1;
        }
#pragma unroll
        for (int t = 0; t < 8; ++t) {
            int n = c * 256 + nh * 128 + t * 16 + nlo;
            float bv = bias2[n];
#pragma unroll
            for (int j = 0; j < 8; ++j)
                out[(size_t)(rowbase + rloc + j) * ODIM + n] = acc[t][j] + bv;
        }
        __syncthreads();
    }
}

// ---------------- launch ----------------
extern "C" void kernel_launch(void* const* d_in, const int* in_sizes, int n_in,
                              void* d_out, int out_size, void* d_ws, size_t ws_size,
                              hipStream_t stream) {
    const float* state  = (const float*)d_in[0];
    const float* action = (const float*)d_in[1];
    const float* W0 = (const float*)d_in[2];
    const float* b0 = (const float*)d_in[3];
    const float* W1 = (const float*)d_in[4];
    const float* b1 = (const float*)d_in[5];
    const float* W2 = (const float*)d_in[6];
    const float* b2 = (const float*)d_in[7];
    const float* D0 = (const float*)d_in[8];
    const float* U0 = (const float*)d_in[9];
    const float* D1 = (const float*)d_in[10];
    const float* U1 = (const float*)d_in[11];
    const float* D2 = (const float*)d_in[12];
    const float* U2 = (const float*)d_in[13];
    float* out = (float*)d_out;

    // workspace: fused bf16 weights
    __bf16* W0e = (__bf16*)d_ws;                 // [256,896]
    __bf16* W1e = W0e + 256 * 896;               // [256,256]
    __bf16* W2e = W1e + 256 * 256;               // [768,256]

    fuse_lora_weights<<<(256 * 896 + 255) / 256, 256, 0, stream>>>(W0, U0, D0, W0e, 256, 896);
    fuse_lora_weights<<<(256 * 256 + 255) / 256, 256, 0, stream>>>(W1, U1, D1, W1e, 256, 256);
    fuse_lora_weights<<<(768 * 256 + 255) / 256, 256, 0, stream>>>(W2, U2, D2, W2e, 768, 256);

    const int nblocks = 65536 / BM;  // 1024 workgroups
    lora_mlp_fused<<<nblocks, NTH, LDS_TOTAL, stream>>>(state, action,
                                                        W0e, b0, W1e, b1, W2e, b2, out);
}
